// WaveletEncoderHybrid_79645873537312
// MI455X (gfx1250) — compile-verified, matrix operand order checked by reading, not run on hardware
//
#include <hip/hip_runtime.h>

typedef __attribute__((ext_vector_type(16))) _Float16 v16h;
typedef __attribute__((ext_vector_type(8)))  _Float16 v8h;
typedef __attribute__((ext_vector_type(8)))  float    v8f;

#define EPSF 1e-5f

union V16 { v16h v; v8h h[2]; };

// ---------------------------------------------------------------------------
// Weight packing: produce the exact WMMA B-matrix register image.
// Layout: wp[((chunk*NT + nt)*32 + lane)*16 + j]
// lane<16: kbase=0 ; lane>=16: kbase=8 ; j<8 -> K=kbase+j ; j>=8 -> K=kbase+8+j
// K ordering: k = tap*CinPad + cin, tap = ky*3+kx. cin >= CinLog (padding) -> 0.
// ---------------------------------------------------------------------------
__global__ void pack_weights_kernel(const float* __restrict__ wsrc,
                                    _Float16* __restrict__ wp,
                                    int Cout, int CinPad, int CinLog, int nChunks)
{
    int idx = blockIdx.x * blockDim.x + threadIdx.x;
    int NTl = Cout >> 4;
    int total = nChunks * NTl * 512;
    if (idx >= total) return;
    int j    = idx & 15;
    int lane = (idx >> 4) & 31;
    int rest = idx >> 9;
    int nt   = rest % NTl;
    int ch   = rest / NTl;
    int kb   = (lane < 16) ? 0 : 8;
    int k    = ch * 32 + kb + (j < 8 ? j : 8 + j);
    int cout = nt * 16 + (lane & 15);
    int tap  = k / CinPad;
    int cin  = k - tap * CinPad;
    float v = 0.f;
    if (tap < 9 && cin < CinLog) {
        int ky = tap / 3, kx = tap - (tap / 3) * 3;
        v = wsrc[(((size_t)cout * CinLog + cin) * 3 + ky) * 3 + kx];
    }
    wp[idx] = (_Float16)v;
}

// scale/shift prep: relu((acc + b)*s + bnb) == relu(acc*s + (b*s + bnb))
__global__ void scale_shift_kernel(const float* __restrict__ g,
                                   const float* __restrict__ bconv,
                                   const float* __restrict__ bnb,
                                   float* __restrict__ sc, float* __restrict__ sh, int C)
{
    int i = blockIdx.x * blockDim.x + threadIdx.x;
    if (i >= C) return;
    float s = g[i] * rsqrtf(1.f + EPSF);
    sc[i] = s;
    sh[i] = bconv[i] * s + bnb[i];
}

// ---------------------------------------------------------------------------
// Implicit-GEMM 3x3 conv with WMMA f16, fused scale/shift/ReLU epilogue.
// Block = 256 threads (8 waves); block tile = 128*MT pixels; wave tile =
// MT x 16 pixels x NT*16 output channels (B fragments reused across MT).
// LDS A tile is double-buffered: one barrier per K-chunk.
// CINP = activation channel stride (padded); CINLOG = logical Cin.
// CINP % 32 == 0 enables the vector-copy staging fast path (one tap per chunk).
// ---------------------------------------------------------------------------
template<int CINP, int CINLOG, int NT, int MT, bool ZEROPAD, bool OUT_NCHW>
__global__ __launch_bounds__(256)
void conv3x3_wmma(const _Float16* __restrict__ act, int Bn, int Hh, int Ww,
                  const _Float16* __restrict__ wpack,
                  const float* __restrict__ scale, const float* __restrict__ shift,
                  _Float16* __restrict__ outH, int outStride, int outOffset,
                  float* __restrict__ outF)
{
    constexpr bool FAST = (CINP % 32) == 0;
    constexpr int  CPT  = FAST ? (CINP / 32) : 1;          // chunks per tap
    constexpr int  NCH  = FAST ? CPT * 9 : (CINLOG * 9 + 31) / 32;
    constexpr int  KTOT = CINLOG * 9;
    constexpr int  ROWS = 128 * MT;

    __shared__ __attribute__((aligned(16))) _Float16 lA[2][ROWS * 40];

    const int t    = threadIdx.x;
    const int lane = t & 31;
    const int wid  = t >> 5;
    const int tileP = blockIdx.x * ROWS;
    const int hw = Hh * Ww;

    // staging coordinates: two threads per pixel row, 16 K-halfs each;
    // each thread stages MT rows (pl, pl+128, ...)
    const int pl   = t >> 1;
    const int kb2  = (t & 1) * 16;
    int b_s[MT], y_s[MT], x_s[MT];
    #pragma unroll
    for (int mi = 0; mi < MT; ++mi) {
        int p_s = tileP + pl + mi * 128;
        int b   = p_s / hw;
        int rem = p_s - b * hw;
        b_s[mi] = b;
        y_s[mi] = rem / Ww;
        x_s[mi] = rem - (rem / Ww) * Ww;
    }

    v8f acc[MT][NT] = {};

    auto stage = [&](int ch, _Float16* buf) {
        if constexpr (FAST) {
            int tap = ch / CPT;
            int sub = ch - tap * CPT;
            int cinBase = sub * 32 + kb2;
            int dy = tap / 3 - 1;
            int dx = tap - (tap / 3) * 3 - 1;
            #pragma unroll
            for (int mi = 0; mi < MT; ++mi) {
                int yy = y_s[mi] + dy, xx = x_s[mi] + dx;
                bool oob = false;
                if (ZEROPAD) {
                    oob = (yy < 0) || (yy >= Hh) || (xx < 0) || (xx >= Ww);
                    if (oob) { yy = 0; xx = 0; }
                } else {
                    yy = yy < 0 ? -yy : (yy >= Hh ? 2 * Hh - 2 - yy : yy);
                    xx = xx < 0 ? -xx : (xx >= Ww ? 2 * Ww - 2 - xx : xx);
                }
                const v8h* src = (const v8h*)(act +
                    ((size_t)(b_s[mi] * Hh + yy) * Ww + xx) * CINP + cinBase);
                v8h a0 = src[0];
                v8h a1 = src[1];
                if (ZEROPAD && oob) {
                    v8h z = {};
                    a0 = z; a1 = z;
                }
                v8h* dst = (v8h*)(buf + (pl + mi * 128) * 40 + kb2);
                dst[0] = a0;
                dst[1] = a1;
            }
        } else {
            // generic gather (small Cin only); CINLOG is compile-time so the
            // divisions fold to magic multiplies
            #pragma unroll
            for (int mi = 0; mi < MT; ++mi) {
                #pragma unroll
                for (int kk = 0; kk < 16; ++kk) {
                    int k = ch * 32 + kb2 + kk;
                    _Float16 v = (_Float16)0.f;
                    if (k < KTOT) {
                        int tap = k / CINLOG;
                        int cin = k - tap * CINLOG;
                        int dy = tap / 3 - 1;
                        int dx = tap - (tap / 3) * 3 - 1;
                        int yy = y_s[mi] + dy, xx = x_s[mi] + dx;
                        if (ZEROPAD) {
                            if (yy >= 0 && yy < Hh && xx >= 0 && xx < Ww)
                                v = act[((size_t)(b_s[mi] * Hh + yy) * Ww + xx) * CINP + cin];
                        } else {
                            yy = yy < 0 ? -yy : (yy >= Hh ? 2 * Hh - 2 - yy : yy);
                            xx = xx < 0 ? -xx : (xx >= Ww ? 2 * Ww - 2 - xx : xx);
                            v = act[((size_t)(b_s[mi] * Hh + yy) * Ww + xx) * CINP + cin];
                        }
                    }
                    buf[(pl + mi * 128) * 40 + kb2 + kk] = v;
                }
            }
        }
    };

    stage(0, lA[0]);
    __syncthreads();

    for (int ch = 0; ch < NCH; ++ch) {
        _Float16* cur = lA[ch & 1];
        if (ch + 1 < NCH)
            stage(ch + 1, lA[(ch + 1) & 1]);

        // All B fragments up front (coalesced b128 pairs); reused across MT
        V16 bf[NT];
        #pragma unroll
        for (int nt = 0; nt < NT; ++nt) {
            const v8h* gw = (const v8h*)(wpack + (((size_t)ch * NT + nt) * 32 + lane) * 16);
            bf[nt].h[0] = gw[0];
            bf[nt].h[1] = gw[1];
        }
        #pragma unroll
        for (int mi = 0; mi < MT; ++mi) {
            // A fragment: two aligned 16B LDS reads matching the 16-bit A layout
            V16 af;
            int row = wid * (16 * MT) + mi * 16 + (lane & 15);
            int s0 = (lane < 16) ? 0 : 1;
            const v8h* lv = (const v8h*)(cur + row * 40);
            af.h[0] = lv[s0];
            af.h[1] = lv[s0 + 2];
            #pragma unroll
            for (int nt = 0; nt < NT; ++nt)
                acc[mi][nt] = __builtin_amdgcn_wmma_f32_16x16x32_f16(
                    false, af.v, false, bf[nt].v, (short)0, acc[mi][nt], false, false);
        }
        __syncthreads();
    }

    // epilogue: acc*scale + shift, ReLU
    const int col = lane & 15;
    const int mhi = (lane < 16) ? 0 : 8;
    #pragma unroll
    for (int mi = 0; mi < MT; ++mi) {
        #pragma unroll
        for (int nt = 0; nt < NT; ++nt) {
            int n = nt * 16 + col;
            float sc = scale ? scale[n] : 1.f;
            float sh = shift[n];
            #pragma unroll
            for (int r = 0; r < 8; ++r) {
                int m = mhi + r;
                int p = tileP + wid * (16 * MT) + mi * 16 + m;
                float v = acc[mi][nt][r] * sc + sh;
                v = v > 0.f ? v : 0.f;
                if (OUT_NCHW) {
                    int b = p / hw; int rem = p - b * hw;
                    int y = rem / Ww; int x = rem - y * Ww;
                    outF[(((size_t)b * (NT * 16) + n) * Hh + y) * Ww + x] = v;
                } else {
                    outH[(size_t)p * outStride + outOffset + n] = (_Float16)v;
                }
            }
        }
    }
}

// ---------------------------------------------------------------------------
// Haar DWT (stride-2 depthwise): NCHW f32 in -> ll (NCHW f32, NHWC f32, NHWC
// f16) + hf (NHWC f32, channel = cin*3 + {LH,HL,HH})
// ---------------------------------------------------------------------------
__global__ void dwt_kernel(const float* __restrict__ in, int Bn, int h, int w,
                           float* __restrict__ ll_nchw, float* __restrict__ ll_nhwc,
                           _Float16* __restrict__ ll_h, float* __restrict__ hf_nhwc)
{
    int p = blockIdx.x * blockDim.x + threadIdx.x;
    int P = Bn * h * w;
    if (p >= P) return;
    int b = p / (h * w);
    int rem = p - b * (h * w);
    int y = rem / w;
    int x = rem - y * w;
    int H2 = 2 * h, W2 = 2 * w;
    #pragma unroll
    for (int c = 0; c < 3; ++c) {
        size_t base = (((size_t)b * 3 + c) * H2 + 2 * y) * W2 + 2 * x;
        float a  = in[base],      bb = in[base + 1];
        float cc = in[base + W2], dd = in[base + W2 + 1];
        float ll = (a + bb + cc + dd) * 0.5f;
        float lh = (a - bb + cc - dd) * 0.5f;
        float hl = (a + bb - cc - dd) * 0.5f;
        float hh = (a - bb - cc + dd) * 0.5f;
        ll_nchw[(((size_t)b * 3 + c) * h + y) * w + x] = ll;
        ll_nhwc[(size_t)p * 3 + c] = ll;
        ll_h[(size_t)p * 3 + c] = (_Float16)ll;
        hf_nhwc[(size_t)p * 9 + c * 3 + 0] = lh;
        hf_nhwc[(size_t)p * 9 + c * 3 + 1] = hl;
        hf_nhwc[(size_t)p * 9 + c * 3 + 2] = hh;
    }
}

// Per-pixel 3-token attention (bd=3, heads=3, hd=1). out = f16(hf + res*proj(y))
__global__ void hf_attn_kernel(const float* __restrict__ hf,
                               const float* __restrict__ qkvw,
                               const float* __restrict__ projw,
                               const float* __restrict__ projb,
                               const float* __restrict__ resp,
                               int P, _Float16* __restrict__ out)
{
    int p = blockIdx.x * blockDim.x + threadIdx.x;
    if (p >= P) return;
    float x[3][3];
    #pragma unroll
    for (int s = 0; s < 3; ++s)
        #pragma unroll
        for (int d = 0; d < 3; ++d)
            x[s][d] = hf[(size_t)p * 9 + s * 3 + d];
    float q[3][3], k[3][3], v[3][3];
    #pragma unroll
    for (int s = 0; s < 3; ++s)
        #pragma unroll
        for (int o = 0; o < 9; ++o) {
            float val = x[s][0] * qkvw[o * 3] + x[s][1] * qkvw[o * 3 + 1] + x[s][2] * qkvw[o * 3 + 2];
            if (o < 3) q[s][o] = val; else if (o < 6) k[s][o - 3] = val; else v[s][o - 6] = val;
        }
    float y[3][3];
    #pragma unroll
    for (int h = 0; h < 3; ++h)
        #pragma unroll
        for (int sq = 0; sq < 3; ++sq) {
            float s0 = q[sq][h] * k[0][h], s1 = q[sq][h] * k[1][h], s2 = q[sq][h] * k[2][h];
            float mx = fmaxf(fmaxf(s0, s1), s2);
            float e0 = __expf(s0 - mx), e1 = __expf(s1 - mx), e2 = __expf(s2 - mx);
            float sum = e0 + e1 + e2;
            y[sq][h] = (e0 * v[0][h] + e1 * v[1][h] + e2 * v[2][h]) / sum;
        }
    float res = resp[0];
    #pragma unroll
    for (int s = 0; s < 3; ++s)
        #pragma unroll
        for (int o = 0; o < 3; ++o) {
            float pr = y[s][0] * projw[o * 3] + y[s][1] * projw[o * 3 + 1] + y[s][2] * projw[o * 3 + 2] + projb[o];
            out[(size_t)p * 9 + s * 3 + o] = (_Float16)(x[s][o] + res * pr);
        }
}

// Per-pixel 4-token attention: token0 = ll, tokens 1..3 = hf triplets.
// Emits f16 tokens 1..3 (hf_fb_raw, 9 channels).
__global__ void fb_attn_kernel(const float* __restrict__ ll, const float* __restrict__ hf,
                               const float* __restrict__ qkvw,
                               const float* __restrict__ projw,
                               const float* __restrict__ projb,
                               const float* __restrict__ resp,
                               int P, _Float16* __restrict__ out)
{
    int p = blockIdx.x * blockDim.x + threadIdx.x;
    if (p >= P) return;
    float x[4][3];
    #pragma unroll
    for (int d = 0; d < 3; ++d) x[0][d] = ll[(size_t)p * 3 + d];
    #pragma unroll
    for (int i = 0; i < 3; ++i)
        #pragma unroll
        for (int d = 0; d < 3; ++d)
            x[1 + i][d] = hf[(size_t)p * 9 + i * 3 + d];
    float q[4][3], k[4][3], v[4][3];
    #pragma unroll
    for (int s = 0; s < 4; ++s)
        #pragma unroll
        for (int o = 0; o < 9; ++o) {
            float val = x[s][0] * qkvw[o * 3] + x[s][1] * qkvw[o * 3 + 1] + x[s][2] * qkvw[o * 3 + 2];
            if (o < 3) q[s][o] = val; else if (o < 6) k[s][o - 3] = val; else v[s][o - 6] = val;
        }
    float y[4][3];
    #pragma unroll
    for (int h = 0; h < 3; ++h)
        #pragma unroll
        for (int sq = 0; sq < 4; ++sq) {
            float sc[4]; float mx = -1e30f;
            #pragma unroll
            for (int sk = 0; sk < 4; ++sk) { sc[sk] = q[sq][h] * k[sk][h]; mx = fmaxf(mx, sc[sk]); }
            float e[4]; float sum = 0.f;
            #pragma unroll
            for (int sk = 0; sk < 4; ++sk) { e[sk] = __expf(sc[sk] - mx); sum += e[sk]; }
            float a = 0.f;
            #pragma unroll
            for (int sk = 0; sk < 4; ++sk) a += e[sk] * v[sk][h];
            y[sq][h] = a / sum;
        }
    float res = resp[0];
    #pragma unroll
    for (int s = 1; s < 4; ++s)
        #pragma unroll
        for (int o = 0; o < 3; ++o) {
            float pr = y[s][0] * projw[o * 3] + y[s][1] * projw[o * 3 + 1] + y[s][2] * projw[o * 3 + 2] + projb[o];
            out[(size_t)p * 9 + (s - 1) * 3 + o] = (_Float16)(x[s][o] + res * pr);
        }
}

// sigmoid gate (1x1 conv over 128 ch) + blend; fully vectorized (v8h).
__global__ void gate_kernel(const _Float16* __restrict__ f, const _Float16* __restrict__ fb,
                            const float* __restrict__ gw, const float* __restrict__ gb,
                            _Float16* __restrict__ out, int stride, int offc, int P)
{
    int p = blockIdx.x * blockDim.x + threadIdx.x;
    if (p >= P) return;
    const v8h* fv = (const v8h*)(f + (size_t)p * 64);
    const v8h* bv = (const v8h*)(fb + (size_t)p * 64);
    v8h fr[8], br[8];
    #pragma unroll
    for (int i = 0; i < 8; ++i) { fr[i] = fv[i]; br[i] = bv[i]; }
    float s = gb[0];
    #pragma unroll
    for (int i = 0; i < 8; ++i)
        #pragma unroll
        for (int j = 0; j < 8; ++j)
            s += (float)fr[i][j] * gw[i * 8 + j] + (float)br[i][j] * gw[64 + i * 8 + j];
    float g = 1.f / (1.f + __expf(-s));
    v8h* o = (v8h*)(out + (size_t)p * stride + offc);
    #pragma unroll
    for (int i = 0; i < 8; ++i) {
        v8h r;
        #pragma unroll
        for (int j = 0; j < 8; ++j)
            r[j] = (_Float16)((float)fr[i][j] * g + (float)br[i][j] * (1.f - g));
        o[i] = r;
    }
}

// 2x2 avgpool of a 64-ch NHWC f16 tensor into the fused (stride-288) buffer.
// One thread per (pixel, 8-channel group): 4x v8h loads, 1x v8h store.
__global__ void avgpool_to_X(const _Float16* __restrict__ in, _Float16* __restrict__ X,
                             int off, int Bn, int Hout, int Wout)
{
    int idx = blockIdx.x * blockDim.x + threadIdx.x;
    int P = Bn * Hout * Wout;
    if (idx >= P * 8) return;
    int g8 = idx & 7;
    int p  = idx >> 3;
    int b = p / (Hout * Wout);
    int rem = p - b * (Hout * Wout);
    int y = rem / Wout;
    int x = rem - y * Wout;
    int Win = Wout * 2;
    size_t basep = ((size_t)(b * Hout * 2 + 2 * y) * Win + 2 * x) * 64 + g8 * 8;
    size_t rowb  = (size_t)Win * 64;
    v8h a = *(const v8h*)(in + basep);
    v8h b2 = *(const v8h*)(in + basep + 64);
    v8h c = *(const v8h*)(in + basep + rowb);
    v8h d = *(const v8h*)(in + basep + rowb + 64);
    v8h r;
    #pragma unroll
    for (int j = 0; j < 8; ++j)
        r[j] = (_Float16)(((float)a[j] + (float)b2[j] + (float)c[j] + (float)d[j]) * 0.25f);
    *(v8h*)(X + (size_t)p * 288 + off + g8 * 8) = r;
}

// mask: 4x4 maxpool ; err: 4x4 avgpool -> X ch 256/257; zero pad ch 258..287
__global__ void mask_err_kernel(const float* __restrict__ mask, const float* __restrict__ err,
                                _Float16* __restrict__ X, int Bn)
{
    int p = blockIdx.x * blockDim.x + threadIdx.x;
    int P = Bn * 128 * 128;
    if (p >= P) return;
    int b = p / (128 * 128);
    int rem = p - b * (128 * 128);
    int y = rem / 128;
    int x = rem - y * 128;
    float m = -1e30f, e = 0.f;
    #pragma unroll
    for (int dy = 0; dy < 4; ++dy)
        #pragma unroll
        for (int dx = 0; dx < 4; ++dx) {
            size_t q = ((size_t)b * 512 + 4 * y + dy) * 512 + 4 * x + dx;
            m = fmaxf(m, mask[q]);
            e += err[q];
        }
    X[(size_t)p * 288 + 256] = (_Float16)m;
    X[(size_t)p * 288 + 257] = (_Float16)(e * (1.f / 16.f));
    #pragma unroll
    for (int c = 258; c < 288; ++c)
        X[(size_t)p * 288 + c] = (_Float16)0.f;
}

// ---------------------------------------------------------------------------
extern "C" void kernel_launch(void* const* d_in, const int* in_sizes, int n_in,
                              void* d_out, int out_size, void* d_ws, size_t ws_size,
                              hipStream_t stream)
{
    (void)in_sizes; (void)n_in; (void)out_size; (void)ws_size;

    const float* rgb          = (const float*)d_in[0];
    const float* init_mask    = (const float*)d_in[1];
    const float* err_map      = (const float*)d_in[2];
    const float* conv_ll_w1   = (const float*)d_in[3];
    const float* conv_ll_b1   = (const float*)d_in[4];
    const float* conv_ll_w2   = (const float*)d_in[5];
    const float* conv_ll_b2   = (const float*)d_in[6];
    const float* conv_ll_bn_g = (const float*)d_in[7];
    const float* conv_ll_bn_b = (const float*)d_in[8];
    const float* conv_hf_w1   = (const float*)d_in[9];
    const float* conv_hf_b1   = (const float*)d_in[10];
    const float* conv_hf_w2   = (const float*)d_in[11];
    const float* conv_hf_b2   = (const float*)d_in[12];
    const float* conv_hf_bn_g = (const float*)d_in[13];
    const float* conv_hf_bn_b = (const float*)d_in[14];
    const float* hf_res       = (const float*)d_in[15];
    const float* hf_qkv_w     = (const float*)d_in[16];
    const float* hf_proj_w    = (const float*)d_in[17];
    const float* hf_proj_b    = (const float*)d_in[18];
    const float* fb_res       = (const float*)d_in[19];
    const float* fb_qkv_w     = (const float*)d_in[20];
    const float* fb_proj_w    = (const float*)d_in[21];
    const float* fb_proj_b    = (const float*)d_in[22];
    const float* gate_w       = (const float*)d_in[23];
    const float* gate_b       = (const float*)d_in[24];
    const float* fuse_w       = (const float*)d_in[25];
    const float* fuse_b       = (const float*)d_in[26];
    const float* fuse_bn_g    = (const float*)d_in[27];
    const float* fuse_bn_b    = (const float*)d_in[28];

    const int P0 = 8 * 256 * 256;  // level-0 pixels
    const int P1 = 8 * 128 * 128;  // level-1 pixels

    char* base = (char*)d_ws;
    size_t off = 0;
    auto A = [&](size_t bytes) -> char* {
        char* p = base + off;
        off += (bytes + 255) & ~(size_t)255;
        return p;
    };

    // packed weights (WMMA B register images)
    _Float16* wp_ll1 = (_Float16*)A((size_t)1  * 4 * 512 * 2);
    _Float16* wp_ll2 = (_Float16*)A((size_t)18 * 4 * 512 * 2);
    _Float16* wp_hf1 = (_Float16*)A((size_t)3  * 4 * 512 * 2);
    _Float16* wp_hf2 = (_Float16*)A((size_t)18 * 4 * 512 * 2);
    _Float16* wp_fu  = (_Float16*)A((size_t)81 * 8 * 512 * 2);
    float* sc_ll = (float*)A(64 * 4);  float* sh_ll = (float*)A(64 * 4);
    float* sc_hf = (float*)A(64 * 4);  float* sh_hf = (float*)A(64 * 4);
    float* sc_fu = (float*)A(128 * 4); float* sh_fu = (float*)A(128 * 4);

    // level-0 tensors
    float*    ll0_nchw = (float*)A((size_t)P0 * 3 * 4);
    float*    ll0_nhwc = (float*)A((size_t)P0 * 3 * 4);
    _Float16* ll0_h    = (_Float16*)A((size_t)P0 * 3 * 2);
    float*    hf0      = (float*)A((size_t)P0 * 9 * 4);
    _Float16* hfh0     = (_Float16*)A((size_t)P0 * 9 * 2);
    _Float16* hffbraw0 = (_Float16*)A((size_t)P0 * 9 * 2);
    _Float16* tbuf     = (_Float16*)A((size_t)P0 * 64 * 2);   // conv1 tmp (both levels)
    _Float16* llf0     = (_Float16*)A((size_t)P0 * 64 * 2);
    _Float16* hff      = (_Float16*)A((size_t)P0 * 64 * 2);   // hf_f (reused level 1)
    _Float16* hffb     = (_Float16*)A((size_t)P0 * 64 * 2);   // hf_fb (reused level 1)
    _Float16* hfcomb0  = (_Float16*)A((size_t)P0 * 64 * 2);
    // level-1 tensors
    float*    ll1_nchw = (float*)A((size_t)P1 * 3 * 4);
    float*    ll1_nhwc = (float*)A((size_t)P1 * 3 * 4);
    _Float16* ll1_h    = (_Float16*)A((size_t)P1 * 3 * 2);
    float*    hf1      = (float*)A((size_t)P1 * 9 * 4);
    _Float16* hfh1     = (_Float16*)A((size_t)P1 * 9 * 2);
    _Float16* hffbraw1 = (_Float16*)A((size_t)P1 * 9 * 2);
    _Float16* X        = (_Float16*)A((size_t)P1 * 288 * 2);  // fused input, stride 288

    // ---- prep: pack weights, fold BN ----
    pack_weights_kernel<<<(1 * 4 * 512 + 255) / 256, 256, 0, stream>>>(conv_ll_w1, wp_ll1, 64, 3, 3, 1);
    pack_weights_kernel<<<(18 * 4 * 512 + 255) / 256, 256, 0, stream>>>(conv_ll_w2, wp_ll2, 64, 64, 64, 18);
    pack_weights_kernel<<<(3 * 4 * 512 + 255) / 256, 256, 0, stream>>>(conv_hf_w1, wp_hf1, 64, 9, 9, 3);
    pack_weights_kernel<<<(18 * 4 * 512 + 255) / 256, 256, 0, stream>>>(conv_hf_w2, wp_hf2, 64, 64, 64, 18);
    pack_weights_kernel<<<(81 * 8 * 512 + 255) / 256, 256, 0, stream>>>(fuse_w, wp_fu, 128, 288, 258, 81);
    scale_shift_kernel<<<1, 64, 0, stream>>>(conv_ll_bn_g, conv_ll_b2, conv_ll_bn_b, sc_ll, sh_ll, 64);
    scale_shift_kernel<<<1, 64, 0, stream>>>(conv_hf_bn_g, conv_hf_b2, conv_hf_bn_b, sc_hf, sh_hf, 64);
    scale_shift_kernel<<<1, 128, 0, stream>>>(fuse_bn_g, fuse_b, fuse_bn_b, sc_fu, sh_fu, 128);

    // ================= level 0 (256x256) =================
    dwt_kernel<<<P0 / 256, 256, 0, stream>>>(rgb, 8, 256, 256, ll0_nchw, ll0_nhwc, ll0_h, hf0);
    hf_attn_kernel<<<P0 / 256, 256, 0, stream>>>(hf0, hf_qkv_w, hf_proj_w, hf_proj_b, hf_res, P0, hfh0);
    fb_attn_kernel<<<P0 / 256, 256, 0, stream>>>(ll0_nhwc, hf0, fb_qkv_w, fb_proj_w, fb_proj_b, fb_res, P0, hffbraw0);

    // ll block
    conv3x3_wmma<3, 3, 4, 2, false, false><<<P0 / 256, 256, 0, stream>>>(
        ll0_h, 8, 256, 256, wp_ll1, nullptr, conv_ll_b1, tbuf, 64, 0, nullptr);
    conv3x3_wmma<64, 64, 4, 2, false, false><<<P0 / 256, 256, 0, stream>>>(
        tbuf, 8, 256, 256, wp_ll2, sc_ll, sh_ll, llf0, 64, 0, nullptr);
    // hf_f block
    conv3x3_wmma<9, 9, 4, 2, false, false><<<P0 / 256, 256, 0, stream>>>(
        hfh0, 8, 256, 256, wp_hf1, nullptr, conv_hf_b1, tbuf, 64, 0, nullptr);
    conv3x3_wmma<64, 64, 4, 2, false, false><<<P0 / 256, 256, 0, stream>>>(
        tbuf, 8, 256, 256, wp_hf2, sc_hf, sh_hf, hff, 64, 0, nullptr);
    // hf_fb block
    conv3x3_wmma<9, 9, 4, 2, false, false><<<P0 / 256, 256, 0, stream>>>(
        hffbraw0, 8, 256, 256, wp_hf1, nullptr, conv_hf_b1, tbuf, 64, 0, nullptr);
    conv3x3_wmma<64, 64, 4, 2, false, false><<<P0 / 256, 256, 0, stream>>>(
        tbuf, 8, 256, 256, wp_hf2, sc_hf, sh_hf, hffb, 64, 0, nullptr);

    gate_kernel<<<P0 / 256, 256, 0, stream>>>(hff, hffb, gate_w, gate_b, hfcomb0, 64, 0, P0);

    // pool level-0 features into fused buffer (channels 0..63 and 128..191)
    avgpool_to_X<<<(P1 * 8) / 256, 256, 0, stream>>>(llf0, X, 0, 8, 128, 128);
    avgpool_to_X<<<(P1 * 8) / 256, 256, 0, stream>>>(hfcomb0, X, 128, 8, 128, 128);

    // ================= level 1 (128x128) =================
    dwt_kernel<<<P1 / 256, 256, 0, stream>>>(ll0_nchw, 8, 128, 128, ll1_nchw, ll1_nhwc, ll1_h, hf1);
    hf_attn_kernel<<<P1 / 256, 256, 0, stream>>>(hf1, hf_qkv_w + 27, hf_proj_w + 9, hf_proj_b + 3, hf_res + 1, P1, hfh1);
    fb_attn_kernel<<<P1 / 256, 256, 0, stream>>>(ll1_nhwc, hf1, fb_qkv_w + 27, fb_proj_w + 9, fb_proj_b + 3, fb_res + 1, P1, hffbraw1);

    // ll block (conv2 writes directly into fused buffer channels 64..127)
    conv3x3_wmma<3, 3, 4, 2, false, false><<<P1 / 256, 256, 0, stream>>>(
        ll1_h, 8, 128, 128, wp_ll1, nullptr, conv_ll_b1, tbuf, 64, 0, nullptr);
    conv3x3_wmma<64, 64, 4, 2, false, false><<<P1 / 256, 256, 0, stream>>>(
        tbuf, 8, 128, 128, wp_ll2, sc_ll, sh_ll, X, 288, 64, nullptr);
    // hf_f block
    conv3x3_wmma<9, 9, 4, 2, false, false><<<P1 / 256, 256, 0, stream>>>(
        hfh1, 8, 128, 128, wp_hf1, nullptr, conv_hf_b1, tbuf, 64, 0, nullptr);
    conv3x3_wmma<64, 64, 4, 2, false, false><<<P1 / 256, 256, 0, stream>>>(
        tbuf, 8, 128, 128, wp_hf2, sc_hf, sh_hf, hff, 64, 0, nullptr);
    // hf_fb block
    conv3x3_wmma<9, 9, 4, 2, false, false><<<P1 / 256, 256, 0, stream>>>(
        hffbraw1, 8, 128, 128, wp_hf1, nullptr, conv_hf_b1, tbuf, 64, 0, nullptr);
    conv3x3_wmma<64, 64, 4, 2, false, false><<<P1 / 256, 256, 0, stream>>>(
        tbuf, 8, 128, 128, wp_hf2, sc_hf, sh_hf, hffb, 64, 0, nullptr);

    // gate writes directly into fused buffer channels 192..255
    gate_kernel<<<P1 / 256, 256, 0, stream>>>(hff, hffb, gate_w + 128, gate_b + 1, X, 288, 192, P1);

    // mask/err 4x4 pools -> channels 256/257 (+ zero the 258..287 padding)
    mask_err_kernel<<<P1 / 256, 256, 0, stream>>>(init_mask, err_map, X, 8);

    // ================= fuse conv (258->128, zero pad, K padded to 288/tap) ==
    conv3x3_wmma<288, 258, 8, 1, true, true><<<P1 / 128, 256, 0, stream>>>(
        X, 8, 128, 128, wp_fu, sc_fu, sh_fu, nullptr, 0, 0, (float*)d_out);
}
